// KANFIS_70033736729042
// MI455X (gfx1250) — matrix operand order repaired and checked by reading, not compiled, not run on hardware
//
#include <hip/hip_runtime.h>
#include <hip/hip_bf16.h>

// ---------------------------------------------------------------------------
// KANFIS forward, MI455X (gfx1250, wave32).
// Memory-bound pipeline; training-mode BatchNorms force multi-pass.
// The 128->20 dense layer runs on v_wmma_f32_16x16x32_f16 (f32 accumulate),
// with the BN1 affine folded into the f16 weight matrix + a bias vector.
// All reductions are deterministic fixed-order partial sums (graph-replay safe).
// ---------------------------------------------------------------------------

#define B_ROWS 131072
#define NTOT   128          // GroupKAN output dim
#define NFIN   20           // fuzzy_in_dim

typedef _Float16 h8  __attribute__((ext_vector_type(8)));
typedef _Float16 h16 __attribute__((ext_vector_type(16)));
typedef float    f8  __attribute__((ext_vector_type(8)));

// workspace layout (bytes)
static constexpr size_t OFF_PROJH = 0;                        // B*128 f16 = 33,554,432
static constexpr size_t OFF_ZPRE  = 33554432;                 // B*20  f32 = 10,485,760
static constexpr size_t OFF_PART1 = OFF_ZPRE + 10485760;      // 1024*256 f32
static constexpr size_t OFF_PART2 = OFF_PART1 + 1048576;      // 2048*64 f32
static constexpr size_t OFF_BPACK = OFF_PART2 + 524288;       // 4096 f16 (WMMA-swizzled 128x32)
static constexpr size_t OFF_C2    = OFF_BPACK + 8192;         // 32 f32
static constexpr size_t OFF_S2T2  = OFF_C2 + 128;             // 64 f32

// ---------------------------------------------------------------------------
// Stage 1: RBF-KAN activations + grouped 8->16 projections. One row/thread.
// Writes proj as f16 rows [B,128] (WMMA A operand precision).
// ---------------------------------------------------------------------------
__global__ __launch_bounds__(256) void k_stage1(
    const float* __restrict__ x, const float* __restrict__ ctr,
    const float* __restrict__ lw, const float* __restrict__ w,
    const float* __restrict__ linw, const float* __restrict__ pW,
    const float* __restrict__ pb, _Float16* __restrict__ projH)
{
    __shared__ float sc[64], sis[64], sw[64], slin[8], spW[1024], spb[128];
    int t = threadIdx.x;
    if (t < 64) { sc[t] = ctr[t]; sis[t] = 1.0f/(__expf(lw[t]) + 1e-6f); sw[t] = w[t]; }
    if (t < 8)  slin[t] = linw[t];
    for (int i = t; i < 1024; i += 256) spW[i] = pW[i];
    if (t < 128) spb[t] = pb[t];
    __syncthreads();

    size_t row = (size_t)blockIdx.x * 256 + t;
    const float4* xr = (const float4*)(x + row * 64);
    h8* pr = (h8*)(projH + row * 128);

    #pragma unroll
    for (int g = 0; g < 8; ++g) {
        float4 xa = xr[g*2], xb = xr[g*2 + 1];
        float xv[8] = {xa.x, xa.y, xa.z, xa.w, xb.x, xb.y, xb.z, xb.w};
        float act[8];
        #pragma unroll
        for (int f = 0; f < 8; ++f) {
            float v = xv[f];
            float s = slin[g] * v;
            #pragma unroll
            for (int k = 0; k < 8; ++k) {
                float d = (v - sc[g*8 + k]) * sis[g*8 + k];
                s += sw[g*8 + k] * __expf(-0.5f * d * d);
            }
            act[f] = s;
        }
        h8 lo, hi;
        #pragma unroll
        for (int o = 0; o < 16; ++o) {
            float acc = spb[g*16 + o];
            #pragma unroll
            for (int f = 0; f < 8; ++f) acc += act[f] * spW[(g*16 + o)*8 + f];
            _Float16 hv = (_Float16)acc;
            if (o < 8) lo[o] = hv; else hi[o - 8] = hv;
        }
        pr[g*2] = lo; pr[g*2 + 1] = hi;
    }
}

// ---------------------------------------------------------------------------
// BN1 stats: per-block partial sum/sumsq of proj over 128 rows (deterministic).
// ---------------------------------------------------------------------------
__global__ __launch_bounds__(256) void k_stats1(
    const _Float16* __restrict__ projH, float* __restrict__ part1)
{
    __shared__ float red[512];
    int t = threadIdx.x;
    int c = t & 127, half = t >> 7;
    size_t base = ((size_t)blockIdx.x * 128 + half * 64) * 128 + c;
    float s = 0.f, q = 0.f;
    #pragma unroll 4
    for (int j = 0; j < 64; ++j) {
        float v = (float)projH[base + (size_t)j * 128];
        s += v; q += v * v;
    }
    red[t] = s; red[256 + t] = q;
    __syncthreads();
    if (t < 128) {
        part1[(size_t)blockIdx.x * 256 + t]       = red[t] + red[t + 128];
        part1[(size_t)blockIdx.x * 256 + 128 + t] = red[256 + t] + red[384 + t];
    }
}

// ---------------------------------------------------------------------------
// Finalize BN1 -> fold affine into f16 weight (WMMA-swizzled Bpack) + bias c2.
// Bpack[pq][lane][idx] = Bele[32*q + 8*(lane>>4) + idx + 8*(idx>>3)][16*p + (lane&15)]
// per ISA 7.12.2 B-matrix layout (column-per-lane, 8-K groups per half-wave).
// ---------------------------------------------------------------------------
__global__ __launch_bounds__(256) void k_finalize1(
    const float* __restrict__ part1, const float* __restrict__ g1,
    const float* __restrict__ b1, const float* __restrict__ fpW,
    const float* __restrict__ fpb, _Float16* __restrict__ Bpack,
    float* __restrict__ c2)
{
    __shared__ float red[256], s1s[128], t1s[128];
    int t = threadIdx.x;
    float acc = 0.f;
    for (int blk = 0; blk < 1024; ++blk) acc += part1[(size_t)blk * 256 + t];
    red[t] = acc;
    __syncthreads();
    if (t < 128) {
        float mu  = red[t] * (1.0f / B_ROWS);
        float var = red[t + 128] * (1.0f / B_ROWS) - mu * mu;
        float s1  = g1[t] * rsqrtf(var + 1e-5f);
        s1s[t] = s1;
        t1s[t] = b1[t] - mu * s1;
    }
    __syncthreads();
    int p = t >> 7, q = (t >> 5) & 3, L = t & 31;
    int col   = 16 * p + (L & 15);
    int kbase = 32 * q + 8 * (L >> 4);
    #pragma unroll
    for (int idx = 0; idx < 16; ++idx) {
        int k = kbase + idx + 8 * (idx >> 3);
        _Float16 v = (_Float16)0.f;
        if (col < NFIN) v = (_Float16)(s1s[k] * fpW[col * 128 + k]);
        Bpack[t * 16 + idx] = v;
    }
    if (t < 32) {
        float cc = 0.f;
        if (t < NFIN) {
            cc = fpb[t];
            for (int k = 0; k < 128; ++k) cc += t1s[k] * fpW[t * 128 + k];
        }
        c2[t] = cc;
    }
}

// ---------------------------------------------------------------------------
// Stage 2: z_pre[B,20] = proj_f16 @ Bmat + c2 via v_wmma_f32_16x16x32_f16.
// One 16-row tile per wave; K=128 -> 4 chunks; N padded 20->32 (2 tiles).
// A layout per ISA: lane m=L&15 holds row m; v[0..7]=K(8*hi+0..7),
// v[8..15]=K(16+8*hi+0..7) within each 32-K chunk (two contiguous 16B loads).
// ---------------------------------------------------------------------------
__global__ __launch_bounds__(256) void k_stage2(
    const _Float16* __restrict__ projH, const _Float16* __restrict__ Bpack,
    const float* __restrict__ c2, float* __restrict__ zpre)
{
    int t = threadIdx.x;
    int wave = t >> 5, lane = t & 31;
    int m = lane & 15, hi = lane >> 4;
    size_t row0 = ((size_t)blockIdx.x * 8 + wave) * 16;

    h16 Bv[8];
    #pragma unroll
    for (int pq = 0; pq < 8; ++pq) {
        const h8* bp = (const h8*)(Bpack + ((size_t)pq * 32 + lane) * 16);
        Bv[pq] = __builtin_shufflevector(bp[0], bp[1],
                 0,1,2,3,4,5,6,7,8,9,10,11,12,13,14,15);
    }

    const _Float16* arow = projH + (row0 + m) * 128 + 8 * hi;
    f8 C0 = {0.f,0.f,0.f,0.f,0.f,0.f,0.f,0.f};
    f8 C1 = {0.f,0.f,0.f,0.f,0.f,0.f,0.f,0.f};
    #pragma unroll
    for (int q = 0; q < 4; ++q) {
        const h8* a0 = (const h8*)(arow + 32 * q);
        const h8* a1 = (const h8*)(arow + 32 * q + 16);
        h16 A = __builtin_shufflevector(a0[0], a1[0],
                0,1,2,3,4,5,6,7,8,9,10,11,12,13,14,15);
        C0 = __builtin_amdgcn_wmma_f32_16x16x32_f16(false, A, false, Bv[q],
                                                    (short)0, C0, false, false);
        C1 = __builtin_amdgcn_wmma_f32_16x16x32_f16(false, A, false, Bv[4 + q],
                                                    (short)0, C1, false, false);
    }

    // D layout: v8f element r = row (r + 8*hi), col = lane&15 (tile0), 16+lane&15 (tile1)
    int n = lane & 15;
    float ca = c2[n], cb = c2[16 + n];
    #pragma unroll
    for (int r = 0; r < 8; ++r) {
        size_t row = row0 + r + 8 * hi;
        zpre[row * 20 + n] = C0[r] + ca;
        if (n < 4) zpre[row * 20 + 16 + n] = C1[r] + cb;
    }
}

// ---------------------------------------------------------------------------
// BN2 stats over z_pre (deterministic partials).
// ---------------------------------------------------------------------------
__global__ __launch_bounds__(256) void k_stats2(
    const float* __restrict__ zpre, float* __restrict__ part2)
{
    __shared__ float red[512];
    int t = threadIdx.x;
    int c = t & 31, sub = t >> 5;
    float s = 0.f, q = 0.f;
    if (c < NFIN) {
        size_t r0 = (size_t)blockIdx.x * 64 + sub * 8;
        #pragma unroll
        for (int j = 0; j < 8; ++j) {
            float v = zpre[(r0 + j) * 20 + c];
            s += v; q += v * v;
        }
    }
    red[t] = s; red[256 + t] = q;
    __syncthreads();
    if (t < 32) {
        float ss = 0.f, qq = 0.f;
        #pragma unroll
        for (int sb = 0; sb < 8; ++sb) { ss += red[sb*32 + t]; qq += red[256 + sb*32 + t]; }
        part2[(size_t)blockIdx.x * 64 + t]      = ss;
        part2[(size_t)blockIdx.x * 64 + 32 + t] = qq;
    }
}

__global__ __launch_bounds__(64) void k_finalize2(
    const float* __restrict__ part2, const float* __restrict__ g2,
    const float* __restrict__ b2, float* __restrict__ s2t2)
{
    __shared__ float red[64];
    int t = threadIdx.x;
    float acc = 0.f;
    for (int blk = 0; blk < 2048; ++blk) acc += part2[(size_t)blk * 64 + t];
    red[t] = acc;
    __syncthreads();
    if (t < NFIN) {
        float mu  = red[t] * (1.0f / B_ROWS);
        float var = red[32 + t] * (1.0f / B_ROWS) - mu * mu;
        float s2  = g2[t] * rsqrtf(var + 1e-5f);
        s2t2[t]      = s2;
        s2t2[32 + t] = b2[t] - mu * s2;
    }
}

// ---------------------------------------------------------------------------
// Stage 3: BN2 affine -> exact GELU -> IT2 fuzzy memberships -> head.
// ---------------------------------------------------------------------------
__global__ __launch_bounds__(256) void k_stage3(
    const float* __restrict__ zpre, const float* __restrict__ s2t2,
    const float* __restrict__ fzc, const float* __restrict__ lsu,
    const float* __restrict__ lsl, const float* __restrict__ hW,
    const float* __restrict__ hb, float* __restrict__ out)
{
    __shared__ float s2s[32], t2s[32], cc[200], isu[200], isl[200], shw[10], shb;
    int t = threadIdx.x;
    if (t < 32) { s2s[t] = s2t2[t]; t2s[t] = s2t2[32 + t]; }
    if (t < 200) {
        cc[t] = fzc[t];
        float su = __expf(lsu[t]) + 1e-6f;
        isu[t] = 1.f / su;
        float sl = fminf(__expf(lsl[t]) + 1e-6f, su * 0.9f);
        isl[t] = 1.f / sl;
    }
    if (t < 10) shw[t] = hW[t];
    if (t == 0) shb = hb[0];
    __syncthreads();

    size_t row = (size_t)blockIdx.x * 256 + t;
    const float* zr = zpre + row * 20;
    float z[20];
    #pragma unroll
    for (int j = 0; j < 20; ++j) {
        float v = zr[j] * s2s[j] + t2s[j];
        z[j] = 0.5f * v * (1.0f + erff(v * 0.70710678118654752f)); // exact GELU
    }
    float acc = shb;
    #pragma unroll
    for (int r = 0; r < 10; ++r) {
        float u = 0.f, l = 0.f;
        #pragma unroll
        for (int j = 0; j < 20; ++j) {
            float d  = z[j] - cc[r*20 + j];
            float du = d * isu[r*20 + j];
            float dl = d * isl[r*20 + j];
            u += __expf(-0.5f * du * du);
            l += __expf(-0.5f * dl * dl);
        }
        acc += shw[r] * (0.5f * (u + l) * (1.0f / 20.0f));
    }
    out[row] = acc;
}

// ---------------------------------------------------------------------------
extern "C" void kernel_launch(void* const* d_in, const int* in_sizes, int n_in,
                              void* d_out, int out_size, void* d_ws, size_t ws_size,
                              hipStream_t stream)
{
    (void)in_sizes; (void)n_in; (void)out_size; (void)ws_size;
    const float* x    = (const float*)d_in[0];
    const float* ctr  = (const float*)d_in[1];
    const float* lw   = (const float*)d_in[2];
    const float* rw   = (const float*)d_in[3];
    const float* linw = (const float*)d_in[4];
    const float* pW   = (const float*)d_in[5];
    const float* pb   = (const float*)d_in[6];
    const float* g1   = (const float*)d_in[7];
    const float* b1   = (const float*)d_in[8];
    const float* fpW  = (const float*)d_in[9];
    const float* fpb  = (const float*)d_in[10];
    const float* g2   = (const float*)d_in[11];
    const float* b2   = (const float*)d_in[12];
    const float* fzc  = (const float*)d_in[13];
    const float* lsu  = (const float*)d_in[14];
    const float* lsl  = (const float*)d_in[15];
    const float* hW   = (const float*)d_in[16];
    const float* hb   = (const float*)d_in[17];

    char* ws = (char*)d_ws;
    _Float16* projH = (_Float16*)(ws + OFF_PROJH);
    float*    zpre  = (float*)(ws + OFF_ZPRE);
    float*    part1 = (float*)(ws + OFF_PART1);
    float*    part2 = (float*)(ws + OFF_PART2);
    _Float16* Bpack = (_Float16*)(ws + OFF_BPACK);
    float*    c2    = (float*)(ws + OFF_C2);
    float*    s2t2  = (float*)(ws + OFF_S2T2);

    k_stage1   <<< B_ROWS/256, 256, 0, stream >>>(x, ctr, lw, rw, linw, pW, pb, projH);
    k_stats1   <<< B_ROWS/128, 256, 0, stream >>>(projH, part1);
    k_finalize1<<< 1,          256, 0, stream >>>(part1, g1, b1, fpW, fpb, Bpack, c2);
    k_stage2   <<< B_ROWS/128, 256, 0, stream >>>(projH, Bpack, c2, zpre);
    k_stats2   <<< B_ROWS/64,  256, 0, stream >>>(zpre, part2);
    k_finalize2<<< 1,           64, 0, stream >>>(part2, g2, b2, s2t2);
    k_stage3   <<< B_ROWS/256, 256, 0, stream >>>(zpre, s2t2, fzc, lsu, lsl, hW, hb,
                                                  (float*)d_out);
}
